// SimpleLowRankLayer_19954418057628
// MI455X (gfx1250) — compile-verified, matrix operand order checked by reading, not run on hardware
//
#include <hip/hip_runtime.h>

#define IN_F  4096
#define OUT_F 4096
#define RANK  16
#define BM    128
#define BN    128
#define BK    32
#define LSTR  40   // padded LDS row stride in bf16 elements (80 bytes -> conflict-free b128)

#define USE_ASYNC_LDS 1

typedef __attribute__((ext_vector_type(16))) __bf16 v16bf;
typedef __attribute__((ext_vector_type(8)))  __bf16 v8bf;
typedef __attribute__((ext_vector_type(4)))  __bf16 v4bf;
typedef __attribute__((ext_vector_type(8)))  float  v8f;
typedef __attribute__((ext_vector_type(4)))  float  v4f;

// ---------------------------------------------------------------------------
// Kernel 1: W_eff = W + B @ A, converted to bf16 into workspace.
// grid = (IN_F/(256*4), OUT_F), block = 256. Tiny (0.5 GFLOP, ~100 MB traffic).
// B row is block-uniform -> compiler can keep it in SGPRs (s_load).
// ---------------------------------------------------------------------------
__global__ __launch_bounds__(256) void build_weff_bf16(
    const float* __restrict__ W, const float* __restrict__ A,
    const float* __restrict__ Bm, __bf16* __restrict__ Weff) {
  const int n  = blockIdx.y;
  const int k4 = (blockIdx.x * 256 + threadIdx.x) * 4;

  float br[RANK];
#pragma unroll
  for (int r = 0; r < RANK; ++r) br[r] = Bm[n * RANK + r];

  v4f acc = *(const v4f*)(W + (size_t)n * IN_F + k4);
#pragma unroll
  for (int r = 0; r < RANK; ++r) {
    v4f a = *(const v4f*)(A + (size_t)r * IN_F + k4);
#pragma unroll
    for (int i = 0; i < 4; ++i) acc[i] += br[r] * a[i];
  }
  v4bf o;
#pragma unroll
  for (int i = 0; i < 4; ++i) o[i] = (__bf16)acc[i];
  *(v4bf*)(Weff + (size_t)n * IN_F + k4) = o;
}

// ---------------------------------------------------------------------------
// Staging helpers
// ---------------------------------------------------------------------------
__device__ __forceinline__ void stage_x(__bf16* dst, const float* __restrict__ X,
                                        int m0, int kk, int tid) {
  // 128 rows x 32 fp32 = 1024 float4 chunks, 4 per thread; convert to bf16.
#pragma unroll
  for (int i = 0; i < 4; ++i) {
    const int c = tid + i * 256;
    const int row = c >> 3, col = c & 7;
    v4f v = *(const v4f*)(X + (size_t)(m0 + row) * IN_F + (kk + col * 4));
    v4bf o;
#pragma unroll
    for (int j = 0; j < 4; ++j) o[j] = (__bf16)v[j];
    *(v4bf*)(dst + row * LSTR + col * 4) = o;  // 8B aligned ds_store_b64
  }
}

__device__ __forceinline__ void stage_w(__bf16* dst, const __bf16* __restrict__ Weff,
                                        int n0, int kk, int tid) {
  // 128 rows x 32 bf16 = 512 x 16B chunks, 2 per thread.
#pragma unroll
  for (int i = 0; i < 2; ++i) {
    const int c = tid + i * 256;
    const int row = c >> 2, col = c & 3;
    const __bf16* src = Weff + (size_t)(n0 + row) * IN_F + kk + col * 8;
    __bf16* d = dst + row * LSTR + col * 8;
#if USE_ASYNC_LDS
    unsigned int loff = (unsigned int)(unsigned long long)(uintptr_t)d;  // low 32 bits = LDS addr
    asm volatile("global_load_async_to_lds_b128 %0, %1, off"
                 :: "v"(loff), "v"(src) : "memory");
#else
    *(v8bf*)d = *(const v8bf*)src;
#endif
  }
}

__device__ __forceinline__ void drain_and_barrier() {
#if USE_ASYNC_LDS
  asm volatile("s_wait_asynccnt 0" ::: "memory");
#endif
  __syncthreads();
}

// 16-bit A/B fragment: lane L holds row (L&15); lanes>=16 take K+8.
// VGPR0-3 <- K[base..base+7], VGPR4-7 <- K[base+16..base+23].
__device__ __forceinline__ v16bf load_frag(const __bf16* tile, int row0, int lane) {
  const int r  = row0 + (lane & 15);
  const int ko = (lane >> 4) << 3;           // 0 or 8 elements
  const __bf16* p = tile + r * LSTR + ko;
  v8bf lo = *(const v8bf*)p;                  // ds_load_b128
  v8bf hi = *(const v8bf*)(p + 16);           // +32 bytes
  return __builtin_shufflevector(lo, hi, 0, 1, 2, 3, 4, 5, 6, 7,
                                         8, 9, 10, 11, 12, 13, 14, 15);
}

// ---------------------------------------------------------------------------
// Kernel 2: out[m,n] = sum_k x[m,k]*Weff[n,k] + b[n]
// block 256 (8 waves), tile 128x128, BK=32, double buffered.
// Wave (wm,wn) in 2x4 grid computes 64x32 = 4x2 WMMA tiles.
// ---------------------------------------------------------------------------
__global__ __launch_bounds__(256) void lora_gemm_bf16(
    const float* __restrict__ X, const __bf16* __restrict__ Weff,
    const float* __restrict__ bias, float* __restrict__ out) {
  __shared__ __bf16 sA[2][BM * LSTR];
  __shared__ __bf16 sB[2][BN * LSTR];

  const int tid  = threadIdx.x;
  const int lane = tid & 31;
  const int wid  = tid >> 5;
  const int wm   = wid >> 2;  // 0..1 -> 64-row slab
  const int wn   = wid & 3;   // 0..3 -> 32-col slab
  const int m0   = blockIdx.y * BM;
  const int n0   = blockIdx.x * BN;

  v8f acc[4][2] = {};

  stage_x(sA[0], X, m0, 0, tid);
  stage_w(sB[0], Weff, n0, 0, tid);
  drain_and_barrier();

  int buf = 0;
  for (int kk = 0; kk < IN_F; kk += BK) {
    if (kk + BK < IN_F) {
      stage_x(sA[buf ^ 1], X, m0, kk + BK, tid);
      stage_w(sB[buf ^ 1], Weff, n0, kk + BK, tid);
    }

    v16bf afr[4], bfr[2];
#pragma unroll
    for (int i = 0; i < 4; ++i) afr[i] = load_frag(sA[buf], wm * 64 + i * 16, lane);
#pragma unroll
    for (int j = 0; j < 2; ++j) bfr[j] = load_frag(sB[buf], wn * 32 + j * 16, lane);

#pragma unroll
    for (int i = 0; i < 4; ++i)
#pragma unroll
      for (int j = 0; j < 2; ++j)
        acc[i][j] = __builtin_amdgcn_wmma_f32_16x16x32_bf16(
            false, afr[i], false, bfr[j], (short)0, acc[i][j], false, false);

    drain_and_barrier();
    buf ^= 1;
  }

  // Epilogue: C/D layout -> VGPR r holds row r (+8 for lanes 16-31), lane&15 = col.
  const int col  = lane & 15;
  const int rsel = (lane >> 4) * 8;
#pragma unroll
  for (int j = 0; j < 2; ++j) {
    const int n = n0 + wn * 32 + j * 16 + col;
    const float bv = bias[n];
#pragma unroll
    for (int i = 0; i < 4; ++i) {
      const int mb = m0 + wm * 64 + i * 16 + rsel;
#pragma unroll
      for (int r = 0; r < 8; ++r)
        out[(size_t)(mb + r) * OUT_F + n] = acc[i][j][r] + bv;
    }
  }
}

// ---------------------------------------------------------------------------
extern "C" void kernel_launch(void* const* d_in, const int* in_sizes, int n_in,
                              void* d_out, int out_size, void* d_ws, size_t ws_size,
                              hipStream_t stream) {
  (void)in_sizes; (void)n_in; (void)out_size; (void)ws_size;
  const float* x  = (const float*)d_in[0];
  const float* W  = (const float*)d_in[1];
  const float* b  = (const float*)d_in[2];
  const float* A  = (const float*)d_in[3];
  const float* Bm = (const float*)d_in[4];
  float*  out  = (float*)d_out;
  __bf16* Weff = (__bf16*)d_ws;  // needs OUT_F*IN_F*2 = 33.5 MB scratch

  dim3 g1(IN_F / (256 * 4), OUT_F);
  build_weff_bf16<<<g1, 256, 0, stream>>>(W, A, Bm, Weff);

  const int M = 4 * 4096;
  dim3 g2(OUT_F / BN, M / BM);
  lora_gemm_bf16<<<g2, 256, 0, stream>>>(x, Weff, b, out);
}
